// MinimalRSNN_42116449305214
// MI455X (gfx1250) — matrix operand order repaired and checked by reading, not compiled, not run on hardware
//
#include <hip/hip_runtime.h>

// ---------------------------------------------------------------------------
// GLIF3/AlphaPSC RSNN for MI455X (gfx1250): single persistent kernel.
//   - 32 workgroups x 128 threads (4 wave32 / WG), each WG owns 32 hidden cols
//   - W_rec / W_in slices staged fp32->f16 into LDS (103.5 KB/WG, pad-strided)
//   - per-step GEMMs via v_wmma_f32_16x16x32_f16, f32 accumulation
//   - neuron state (v, asc0/1, ref, h, psc, rate) fully register-resident
//   - spikes exchanged via double-buffered global f16 buffer (L2 resident),
//     addressed off a single base pointer so loads lower to global_load_b128
//     (not flat_*, which would couple the vector path to DScnt)
//   - one agent-scope atomic grid barrier per step
// ---------------------------------------------------------------------------

typedef __attribute__((ext_vector_type(16))) _Float16 v16h;
typedef __attribute__((ext_vector_type(8)))  _Float16 v8h;
typedef __attribute__((ext_vector_type(4)))  _Float16 v4h;
typedef __attribute__((ext_vector_type(8)))  float    v8f;
typedef __attribute__((ext_vector_type(4)))  float    v4f;

#define T_STEPS   1000
#define BATCH     32
#define NI_DIM    512
#define NH_DIM    1024
#define NO_DIM    256

#define NWG       32          // workgroups (grid barrier participants)
#define COLS      32          // hidden columns per WG
#define NTHREADS  128         // 4 wave32

#define REC_STRIDE  (NH_DIM + 8)   // LDS row stride in halfs (bank-conflict pad)
#define WIN_STRIDE  (NI_DIM + 8)
#define RATE_STRIDE 33

#define V_TH     (-45.0f)
#define V_RESET  (-60.0f)
#define INV_TAU  (0.05f)       // 1/20
#define INV_CM   (0.5f)        // 1/2
#define TAU_REF  (2.0f)
#define ASC_A0   (1.0f)
#define ASC_A1   (-2.0f)
#define DEC_SYN  (0.81873075307798182f)  // exp(-1/5)
#define DEC_A0   (0.90483741803595957f)  // exp(-0.1)
#define DEC_A1   (0.81873075307798182f)  // exp(-0.2)

union ATile { v16h v; v8h h[2]; };

__device__ __forceinline__ v8h cvt8(v4f lo, v4f hi) {
  v8h r;
  r[0] = (_Float16)lo.x; r[1] = (_Float16)lo.y;
  r[2] = (_Float16)lo.z; r[3] = (_Float16)lo.w;
  r[4] = (_Float16)hi.x; r[5] = (_Float16)hi.y;
  r[6] = (_Float16)hi.z; r[7] = (_Float16)hi.w;
  return r;
}

extern "C" __global__ void rsnn_init(float* __restrict__ out, unsigned* __restrict__ bar) {
  int i = blockIdx.x * blockDim.x + threadIdx.x;
  if (i < BATCH * NO_DIM) out[i] = 0.0f;
  if (i == 0) *bar = 0u;
}

extern "C" __global__ void __launch_bounds__(NTHREADS, 1)
rsnn_persistent(const float* __restrict__ x,
                const float* __restrict__ Win,
                const float* __restrict__ Wrec,
                const float* __restrict__ Wout,
                float* __restrict__ out,
                unsigned* __restrict__ bar,
                _Float16* __restrict__ sbuf)   // 2 x [BATCH x NH] f16, contiguous
{
  extern __shared__ char smem[];
  _Float16* ldsWrec = (_Float16*)smem;                              // COLS x REC_STRIDE
  _Float16* ldsWin  = ldsWrec + (size_t)COLS * REC_STRIDE;          // COLS x WIN_STRIDE
  float*    ldsRate = (float*)(ldsWin + (size_t)COLS * WIN_STRIDE); // COLS x RATE_STRIDE

  const int tid   = threadIdx.x;
  const int g     = blockIdx.x;
  const int hbase = g * COLS;

  // ---- stage weight slices (fp32 -> f16) into LDS, vectorized float4 ----
  for (int c = tid; c < COLS * NH_DIM / 4; c += NTHREADS) {
    int e = c * 4;
    int row = e >> 10;            // NH_DIM = 1024
    int col = e & (NH_DIM - 1);
    v4f f = *(const v4f*)(Wrec + (size_t)(hbase + row) * NH_DIM + col);
    v4h hv; hv[0]=(_Float16)f.x; hv[1]=(_Float16)f.y; hv[2]=(_Float16)f.z; hv[3]=(_Float16)f.w;
    *(v4h*)(ldsWrec + (size_t)row * REC_STRIDE + col) = hv;
  }
  for (int c = tid; c < COLS * NI_DIM / 4; c += NTHREADS) {
    int e = c * 4;
    int row = e >> 9;             // NI_DIM = 512
    int col = e & (NI_DIM - 1);
    v4f f = *(const v4f*)(Win + (size_t)(hbase + row) * NI_DIM + col);
    v4h hv; hv[0]=(_Float16)f.x; hv[1]=(_Float16)f.y; hv[2]=(_Float16)f.z; hv[3]=(_Float16)f.w;
    *(v4h*)(ldsWin + (size_t)row * WIN_STRIDE + col) = hv;
  }
  __syncthreads();

  // ---- wave/lane -> tile mapping (ISA WMMA f16 layouts) ----
  const int lane   = tid & 31;
  const int w      = tid >> 5;
  const int mt     = w >> 1;                 // M tile: batch rows 16*mt..+15
  const int nt     = w & 1;                  // N tile: cols 16*nt..+15
  const int arow_r = mt * 16 + (lane & 15);  // A-tile source row (batch)
  const int ksubA  = (lane >> 4) * 8;        // A lane K sub-offset
  const int ksubB  = (lane >> 4) * 8;        // B lane K sub-offset
  const int bbase  = mt * 16 + (lane >> 4) * 8; // C-tile: this lane's batch row base
  const int hl     = nt * 16 + (lane & 15);     // C-tile: this lane's local column
  const int hg     = hbase + hl;

  const _Float16* bWin  = ldsWin  + (size_t)hl * WIN_STRIDE + ksubB;
  const _Float16* bWrec = ldsWrec + (size_t)hl * REC_STRIDE + ksubB;

  // Per-thread fixed offsets inside a spike buffer half (single global base).
  const size_t spk_wr_base = (size_t)bbase * NH_DIM + hg;           // + r*NH per neuron
  const size_t spk_rd_base = (size_t)arow_r * NH_DIM + ksubA;       // A-tile row

  // ---- register-resident neuron state: 8 neurons per thread ----
  float vmem[8], a0[8], a1[8], refr[8], hstate[8], psc[8], rate[8];
#pragma unroll
  for (int r = 0; r < 8; ++r) {
    vmem[r] = V_RESET; a0[r] = 0.f; a1[r] = 0.f; refr[r] = 0.f;
    hstate[r] = 0.f; psc[r] = 0.f; rate[r] = 0.f;
  }

#pragma unroll 1
  for (int t = 0; t < T_STEPS; ++t) {
    const size_t bufoff = (size_t)(t & 1) * ((size_t)BATCH * NH_DIM);

    // ---------- input current: cI = x[t] @ Win^T (own column slice) ----------
    const float* xrow = x + ((size_t)t * BATCH + arow_r) * NI_DIM + ksubA;
    v8f cI0 = {}; v8f cI1 = {};
#pragma unroll
    for (int k = 0; k < NI_DIM; k += 64) {
      ATile a, b;
      a.h[0] = cvt8(*(const v4f*)(xrow + k),      *(const v4f*)(xrow + k + 4));
      a.h[1] = cvt8(*(const v4f*)(xrow + k + 16), *(const v4f*)(xrow + k + 20));
      b.h[0] = *(const v8h*)(bWin + k);
      b.h[1] = *(const v8h*)(bWin + k + 16);
      cI0 = __builtin_amdgcn_wmma_f32_16x16x32_f16(false, a.v, false, b.v,
                                                   (short)0, cI0, false, false);
      a.h[0] = cvt8(*(const v4f*)(xrow + k + 32), *(const v4f*)(xrow + k + 36));
      a.h[1] = cvt8(*(const v4f*)(xrow + k + 48), *(const v4f*)(xrow + k + 52));
      b.h[0] = *(const v8h*)(bWin + k + 32);
      b.h[1] = *(const v8h*)(bWin + k + 48);
      cI1 = __builtin_amdgcn_wmma_f32_16x16x32_f16(false, a.v, false, b.v,
                                                   (short)0, cI1, false, false);
    }

    // ---------- GLIF neuron update + spike (all in registers) ----------
#pragma unroll
    for (int r = 0; r < 8; ++r) {
      float I = cI0[r] + cI1[r] + psc[r];
      a0[r] *= DEC_A0;
      a1[r] *= DEC_A1;
      float vv = vmem[r] + (V_RESET - vmem[r]) * INV_TAU + (I + a0[r] + a1[r]) * INV_CM;
      bool inref = refr[r] > 0.f;
      vv = inref ? V_RESET : vv;
      float s = (!inref && vv >= V_TH) ? 1.f : 0.f;   // hard Heaviside forward value
      vv = (s > 0.f) ? V_RESET : vv;                  // hard reset on spike
      vmem[r] = vv;
      a0[r] += s * ASC_A0;
      a1[r] += s * ASC_A1;
      refr[r] = (s > 0.5f) ? TAU_REF : fmaxf(refr[r] - 1.f, 0.f);
      rate[r] += s;
      sbuf[bufoff + spk_wr_base + (size_t)r * NH_DIM] = (_Float16)s; // publish
    }

    // ---------- publish fence + device-wide barrier (one per step) ----------
    __threadfence();
    __syncthreads();
    if (tid == 0) {
      __hip_atomic_fetch_add(bar, 1u, __ATOMIC_RELEASE, __HIP_MEMORY_SCOPE_AGENT);
      const unsigned target = (unsigned)(t + 1) * (unsigned)NWG;
      while (__hip_atomic_load(bar, __ATOMIC_ACQUIRE, __HIP_MEMORY_SCOPE_AGENT) < target) {
        __builtin_amdgcn_s_sleep(2);
      }
    }
    __syncthreads();
    __threadfence();

    // prefetch next timestep's activation rows while doing the recurrent GEMM
    if (t + 1 < T_STEPS) {
      __builtin_prefetch(xrow + (size_t)BATCH * NI_DIM, 0, 1);
      __builtin_prefetch(xrow + (size_t)BATCH * NI_DIM + 64, 0, 1);
    }

    // ---------- recurrent current: cR = spikes @ Wrec^T (own slice) ----------
    const _Float16* srow = sbuf + bufoff + spk_rd_base;
    v8f cR0 = {}; v8f cR1 = {};
#pragma unroll
    for (int k = 0; k < NH_DIM; k += 64) {
      ATile a, b;
      a.h[0] = *(const v8h*)(srow + k);
      a.h[1] = *(const v8h*)(srow + k + 16);
      b.h[0] = *(const v8h*)(bWrec + k);
      b.h[1] = *(const v8h*)(bWrec + k + 16);
      cR0 = __builtin_amdgcn_wmma_f32_16x16x32_f16(false, a.v, false, b.v,
                                                   (short)0, cR0, false, false);
      a.h[0] = *(const v8h*)(srow + k + 32);
      a.h[1] = *(const v8h*)(srow + k + 48);
      b.h[0] = *(const v8h*)(bWrec + k + 32);
      b.h[1] = *(const v8h*)(bWrec + k + 48);
      cR1 = __builtin_amdgcn_wmma_f32_16x16x32_f16(false, a.v, false, b.v,
                                                   (short)0, cR1, false, false);
    }

    // ---------- two-pole alpha synapse filter ----------
#pragma unroll
    for (int r = 0; r < 8; ++r) {
      hstate[r] = DEC_SYN * hstate[r] + (cR0[r] + cR1[r]);
      psc[r]    = DEC_SYN * psc[r] + hstate[r];   // DT = 1
    }
  }

  // ---------- readout: out += (rate/T) @ Wout^T for this column slice ----------
  const float invT = 1.0f / (float)T_STEPS;
#pragma unroll
  for (int r = 0; r < 8; ++r)
    ldsRate[(size_t)hl * RATE_STRIDE + (bbase + r)] = rate[r] * invT;
  __syncthreads();

  for (int p = tid; p < BATCH * NO_DIM; p += NTHREADS) {
    int o = p & (NO_DIM - 1);
    int b = p >> 8;
    const float* wr = Wout + (size_t)o * NH_DIM + hbase;
    float acc = 0.f;
#pragma unroll 8
    for (int c = 0; c < COLS; ++c)
      acc += ldsRate[(size_t)c * RATE_STRIDE + b] * wr[c];
    atomicAdd(out + (size_t)b * NO_DIM + o, acc);
  }
}

extern "C" void kernel_launch(void* const* d_in, const int* in_sizes, int n_in,
                              void* d_out, int out_size, void* d_ws, size_t ws_size,
                              hipStream_t stream) {
  (void)in_sizes; (void)n_in; (void)out_size; (void)ws_size;

  const float* x    = (const float*)d_in[0];   // [T,B,NI] fp32
  const float* Win  = (const float*)d_in[1];   // [NH,NI]  fp32
  const float* Wrec = (const float*)d_in[2];   // [NH,NH]  fp32
  const float* Wout = (const float*)d_in[3];   // [NO,NH]  fp32
  float* out        = (float*)d_out;           // [B,NO]   fp32

  // workspace layout: [0,4)=barrier ctr, [256,...) = two f16 spike buffers
  unsigned*  bar  = (unsigned*)d_ws;
  _Float16*  sbuf = (_Float16*)((char*)d_ws + 256);   // 2 * BATCH*NH halves

  constexpr size_t LDS_BYTES =
      (size_t)COLS * REC_STRIDE * sizeof(_Float16) +
      (size_t)COLS * WIN_STRIDE * sizeof(_Float16) +
      (size_t)COLS * RATE_STRIDE * sizeof(float);   // 103,552 B

  (void)hipFuncSetAttribute((const void*)rsnn_persistent,
                            hipFuncAttributeMaxDynamicSharedMemorySize,
                            (int)LDS_BYTES);

  rsnn_init<<<(BATCH * NO_DIM + 255) / 256, 256, 0, stream>>>(out, bar);
  rsnn_persistent<<<NWG, NTHREADS, LDS_BYTES, stream>>>(
      x, Win, Wrec, Wout, out, bar, sbuf);
}
